// PoolingModule_33397665694048
// MI455X (gfx1250) — compile-verified, matrix operand order checked by reading, not run on hardware
//
#include <hip/hip_runtime.h>

#define BATCH   4
#define NPTS    8192
#define MSAMP   2048
#define KNBR    128
#define CHUNK   256
#define QT      16
#define RAD2    4.0f

typedef __attribute__((ext_vector_type(16))) _Float16 v16h;
typedef __attribute__((ext_vector_type(8)))  float    v8f;

#if defined(__has_builtin) && __has_builtin(__builtin_amdgcn_s_wait_asynccnt)
#define WAIT_ASYNC(n) __builtin_amdgcn_s_wait_asynccnt(n)
#else
#define WAIT_ASYNC(n) asm volatile("s_wait_asynccnt %0" :: "i"(n) : "memory")
#endif

// ---------------------------------------------------------------------------
// Pack pos(3)++x(3) into padded-8 layouts (f32 for FPS, f16 for WMMA) + |p|^2
// ---------------------------------------------------------------------------
__global__ void pack_kernel(const float* __restrict__ x,
                            const float* __restrict__ pos,
                            float* __restrict__ p6f,
                            _Float16* __restrict__ p6h,
                            float* __restrict__ pn) {
    int g = blockIdx.x * blockDim.x + threadIdx.x;
    if (g >= BATCH * NPTS) return;
    float v[8];
    v[0] = pos[g * 3 + 0]; v[1] = pos[g * 3 + 1]; v[2] = pos[g * 3 + 2];
    v[3] = x[g * 3 + 0];   v[4] = x[g * 3 + 1];   v[5] = x[g * 3 + 2];
    v[6] = 0.0f; v[7] = 0.0f;
    float s = 0.0f;
    #pragma unroll
    for (int d = 0; d < 8; ++d) {
        p6f[(size_t)g * 8 + d] = v[d];
        p6h[(size_t)g * 8 + d] = (_Float16)v[d];
        s += v[d] * v[d];
    }
    pn[g] = s;
}

// ---------------------------------------------------------------------------
// Farthest point sampling: 1 workgroup per cloud, dmin resident in LDS.
// ---------------------------------------------------------------------------
__launch_bounds__(1024)
__global__ void fps_kernel(const float* __restrict__ p6f,
                           int* __restrict__ fpsIdx) {
    const int b = blockIdx.x;
    const int t = threadIdx.x;
    const int lane = t & 31, wid = t >> 5;

    __shared__ float dmin[NPTS];          // 32 KB
    __shared__ float redV[32];
    __shared__ int   redI[32];
    __shared__ float cur[8];

    const float* P = p6f + (size_t)b * NPTS * 8;

    if (t < 8) cur[t] = P[t];             // start point = local index 0
    if (t == 0) fpsIdx[b * MSAMP] = 0;
    __syncthreads();

    #pragma unroll
    for (int k = 0; k < NPTS / 1024; ++k) {
        int i = t + k * 1024;
        const float4* pp = (const float4*)(P + (size_t)i * 8);
        float4 a = pp[0], c = pp[1];
        float d0 = a.x - cur[0], d1 = a.y - cur[1], d2 = a.z - cur[2];
        float d3 = a.w - cur[3], d4 = c.x - cur[4], d5 = c.y - cur[5];
        dmin[i] = d0*d0 + d1*d1 + d2*d2 + d3*d3 + d4*d4 + d5*d5;
    }
    __syncthreads();

    for (int it = 1; it < MSAMP; ++it) {
        float bv = -1.0e38f; int bi = 0x7fffffff;
        #pragma unroll
        for (int k = 0; k < NPTS / 1024; ++k) {
            int i = t + k * 1024;
            float v = dmin[i];
            if (v > bv || (v == bv && i < bi)) { bv = v; bi = i; }
        }
        for (int o = 16; o > 0; o >>= 1) {
            float ov = __shfl_down(bv, o, 32);
            int   oi = __shfl_down(bi, o, 32);
            if (ov > bv || (ov == bv && oi < bi)) { bv = ov; bi = oi; }
        }
        if (lane == 0) { redV[wid] = bv; redI[wid] = bi; }
        __syncthreads();
        if (wid == 0) {
            bv = redV[lane]; bi = redI[lane];
            for (int o = 16; o > 0; o >>= 1) {
                float ov = __shfl_down(bv, o, 32);
                int   oi = __shfl_down(bi, o, 32);
                if (ov > bv || (ov == bv && oi < bi)) { bv = ov; bi = oi; }
            }
            if (lane == 0) {
                fpsIdx[b * MSAMP + it] = bi;
                const float* sp = P + (size_t)bi * 8;
                #pragma unroll
                for (int d = 0; d < 8; ++d) cur[d] = sp[d];
            }
        }
        __syncthreads();
        #pragma unroll
        for (int k = 0; k < NPTS / 1024; ++k) {
            int i = t + k * 1024;
            const float4* pp = (const float4*)(P + (size_t)i * 8);
            float4 a = pp[0], c = pp[1];
            float d0 = a.x - cur[0], d1 = a.y - cur[1], d2 = a.z - cur[2];
            float d3 = a.w - cur[3], d4 = c.x - cur[4], d5 = c.y - cur[5];
            float d  = d0*d0 + d1*d1 + d2*d2 + d3*d3 + d4*d4 + d5*d5;
            dmin[i] = fminf(dmin[i], d);
        }
        __syncthreads();
    }
}

// ---------------------------------------------------------------------------
// Gather sampled queries to f16 (WMMA A operand) + |q|^2
// ---------------------------------------------------------------------------
__global__ void gatherq_kernel(const float* __restrict__ p6f,
                               const int* __restrict__ fpsIdx,
                               _Float16* __restrict__ qh,
                               float* __restrict__ qn) {
    int g = blockIdx.x * blockDim.x + threadIdx.x;
    if (g >= BATCH * MSAMP) return;
    int b = g / MSAMP;
    int idx = fpsIdx[g];
    const float* sp = p6f + ((size_t)b * NPTS + idx) * 8;
    float s = 0.0f;
    #pragma unroll
    for (int d = 0; d < 8; ++d) {
        float v = sp[d];
        qh[(size_t)g * 8 + d] = (_Float16)v;
        s += v * v;
    }
    qn[g] = s;
}

// ---------------------------------------------------------------------------
// Neighbor kernel: WG = (cloud b, 16-query tile). Double-buffered async
// global->LDS staging (ASYNCcnt), d2 tiles via v_wmma_f32_16x16x32_f16,
// per-query top-128 replace-max list in LDS, fused pooling epilogue.
// ---------------------------------------------------------------------------
__launch_bounds__(256)
__global__ void nbr_kernel(const _Float16* __restrict__ p6h,
                           const float* __restrict__ pn,
                           const _Float16* __restrict__ qh,
                           const float* __restrict__ qn,
                           const float* __restrict__ xin,
                           const float* __restrict__ posin,
                           float* __restrict__ dout) {
    const int b    = blockIdx.y;
    const int qt   = blockIdx.x;             // 0..MSAMP/QT-1
    const int t    = threadIdx.x;
    const int lane = t & 31, wid = t >> 5;   // 8 waves
    const int qbase = qt * QT;

    __shared__ _Float16 sQ[QT * 8];
    __shared__ float    sQn[QT];
    __shared__ alignas(16) _Float16 sP[2][CHUNK * 8];   // 8 KB (double buffer)
    __shared__ alignas(16) float    sPn[2][CHUNK];      // 2 KB
    __shared__ float    sD[QT][CHUNK];                  // 16 KB
    __shared__ float    listD[QT][KNBR];                // 8 KB
    __shared__ int      listI[QT][KNBR];                // 8 KB

    if (t < QT * 8) sQ[t] = qh[((size_t)(b * MSAMP + qbase)) * 8 + t];
    if (t < QT)     sQn[t] = qn[b * MSAMP + qbase + t];
    for (int s = t; s < QT * KNBR; s += 256) {
        (&listD[0][0])[s] = 3.0e38f;
        (&listI[0][0])[s] = 0;
    }

    // LDS byte offsets (generic LDS pointer low 32 bits == LDS address)
    const unsigned ldsP[2]  = { (unsigned)(uintptr_t)&sP[0][0],
                                (unsigned)(uintptr_t)&sP[1][0] };
    const unsigned ldsPn[2] = { (unsigned)(uintptr_t)&sPn[0][0],
                                (unsigned)(uintptr_t)&sPn[1][0] };

    // async copy of one chunk: each thread moves 16B of points + 4B of |p|^2
    auto issue_chunk = [&](int cb, int bufIdx) {
        unsigned dstP  = ldsP[bufIdx]  + (unsigned)t * 16u;
        unsigned dstPn = ldsPn[bufIdx] + (unsigned)t * 4u;
        unsigned offP  = (unsigned)(b * NPTS + cb + t) * 16u;
        unsigned offPn = (unsigned)(b * NPTS + cb + t) * 4u;
        asm volatile("global_load_async_to_lds_b128 %0, %1, %2"
                     :: "v"(dstP), "v"(offP), "s"(p6h) : "memory");
        asm volatile("global_load_async_to_lds_b32 %0, %1, %2"
                     :: "v"(dstPn), "v"(offPn), "s"(pn) : "memory");
    };

    issue_chunk(0, 0);                       // prologue copy, ASYNCcnt = 2
    __syncthreads();                         // sQ/lists visible

    // A fragment (16 queries x 32, K=0..5 valid): lanes 0-15 hold row M=lane,
    // halves 0..7 = K0..7; lanes 16-31 (K8..15,K24..31) are zero.
    v16h A;
    #pragma unroll
    for (int e = 0; e < 16; ++e) A[e] = (_Float16)0.0f;
    if (lane < 16) {
        #pragma unroll
        for (int e = 0; e < 8; ++e) A[e] = sQ[lane * 8 + e];
    }

    float curMax0 = 3.0e38f, curMax1 = 3.0e38f;   // per owned query (2/wave)
    const int qown0 = wid * 2;
    const int nch = NPTS / CHUNK;

    for (int c = 0; c < nch; ++c) {
        const int buf = c & 1;
        const int cb  = c * CHUNK;
        if (c + 1 < nch) {
            issue_chunk(cb + CHUNK, buf ^ 1);   // overlap next chunk's DMA
            WAIT_ASYNC(2);                      // in-order: chunk c resident
        } else {
            WAIT_ASYNC(0);
        }
        __syncthreads();                        // chunk visible to all waves

        // each wave computes 2 of 16 16x16 tiles
        #pragma unroll
        for (int tt = 0; tt < 2; ++tt) {
            const int pb = (wid * 2 + tt) * 16;   // column base in chunk
            v16h Bf;
            #pragma unroll
            for (int e = 0; e < 16; ++e) Bf[e] = (_Float16)0.0f;
            if (lane < 16) {
                #pragma unroll
                for (int e = 0; e < 8; ++e) Bf[e] = sP[buf][(pb + lane) * 8 + e];
            }
            v8f cc = {};
            cc = __builtin_amdgcn_wmma_f32_16x16x32_f16(
                     false, A, false, Bf, (short)0, cc, false, false);
            const int col   = pb + (lane & 15);
            const int rbase = (lane >> 4) << 3;
            const float pnc = sPn[buf][col];
            #pragma unroll
            for (int r = 0; r < 8; ++r) {
                int row = rbase + r;
                sD[row][col] = sQn[row] + pnc - 2.0f * cc[r];
            }
        }
        __syncthreads();

        // ---- top-128 maintenance: one wave per owned query ----
        #pragma unroll
        for (int qq = 0; qq < 2; ++qq) {
            const int q = qown0 + qq;
            float cm = (qq == 0) ? curMax0 : curMax1;
            float* LD = listD[q];
            int*   LI = listI[q];
            #pragma unroll
            for (int j = 0; j < CHUNK / 32; ++j) {
                float v = sD[q][j * 32 + lane];
                unsigned long long mask = __ballot(v < cm);
                while (mask) {
                    int src = __ffsll(mask) - 1;
                    mask &= mask - 1;
                    float vv = __shfl(v, src, 32);
                    if (vv < cm) {
                        float mv = -1.0f; int mp = 0;
                        #pragma unroll
                        for (int s = 0; s < KNBR / 32; ++s) {
                            float lv = LD[s * 32 + lane];
                            if (lv > mv) { mv = lv; mp = s * 32 + lane; }
                        }
                        for (int o = 16; o > 0; o >>= 1) {
                            float ov = __shfl_down(mv, o, 32);
                            int   op = __shfl_down(mp, o, 32);
                            if (ov > mv) { mv = ov; mp = op; }
                        }
                        mp = __shfl(mp, 0, 32);
                        if (lane == 0) { LD[mp] = vv; LI[mp] = cb + j * 32 + src; }
                        float nv = -1.0f;
                        #pragma unroll
                        for (int s = 0; s < KNBR / 32; ++s) {
                            float lv = LD[s * 32 + lane];
                            if (lv > nv) nv = lv;
                        }
                        for (int o = 16; o > 0; o >>= 1) {
                            float ov = __shfl_down(nv, o, 32);
                            if (ov > nv) nv = ov;
                        }
                        cm = __shfl(nv, 0, 32);
                    }
                }
            }
            if (qq == 0) curMax0 = cm; else curMax1 = cm;
        }
        __syncthreads();
    }

    // ---- epilogue: masked mean pooling + edge_index + batch ----
    const int mTot = BATCH * MSAMP;
    float* xout = dout;
    float* pout = dout + (size_t)mTot * 3;
    int*   bout = (int*)(dout + (size_t)mTot * 6);
    int*   eout = (int*)(dout + (size_t)mTot * 7);

    #pragma unroll
    for (int qq = 0; qq < 2; ++qq) {
        const int q  = qown0 + qq;
        const int qg = b * MSAMP + qbase + q;
        float* LD = listD[q];
        int*   LI = listI[q];
        float cnt = 0.0f, sx0 = 0, sx1 = 0, sx2 = 0, sp0 = 0, sp1 = 0, sp2 = 0;
        const size_t ebase = (size_t)qg * KNBR;
        #pragma unroll
        for (int s = 0; s < KNBR / 32; ++s) {
            int   slot = s * 32 + lane;
            float dd = LD[slot];
            int   ii = LI[slot];
            bool  ok = dd <= RAD2;
            int gidx = b * NPTS + ii;
            eout[ebase + slot] = ok ? gidx : -1;
            eout[(size_t)mTot * KNBR + ebase + slot] = qg;
            if (ok) {
                cnt += 1.0f;
                sx0 += xin[(size_t)gidx * 3 + 0];
                sx1 += xin[(size_t)gidx * 3 + 1];
                sx2 += xin[(size_t)gidx * 3 + 2];
                sp0 += posin[(size_t)gidx * 3 + 0];
                sp1 += posin[(size_t)gidx * 3 + 1];
                sp2 += posin[(size_t)gidx * 3 + 2];
            }
        }
        for (int o = 16; o > 0; o >>= 1) {
            cnt += __shfl_down(cnt, o, 32);
            sx0 += __shfl_down(sx0, o, 32);
            sx1 += __shfl_down(sx1, o, 32);
            sx2 += __shfl_down(sx2, o, 32);
            sp0 += __shfl_down(sp0, o, 32);
            sp1 += __shfl_down(sp1, o, 32);
            sp2 += __shfl_down(sp2, o, 32);
        }
        if (lane == 0) {
            float inv = 1.0f / cnt;                  // cnt >= 1 (self neighbor)
            xout[(size_t)qg * 3 + 0] = sx0 * inv;
            xout[(size_t)qg * 3 + 1] = sx1 * inv;
            xout[(size_t)qg * 3 + 2] = sx2 * inv;
            pout[(size_t)qg * 3 + 0] = sp0 * inv;
            pout[(size_t)qg * 3 + 1] = sp1 * inv;
            pout[(size_t)qg * 3 + 2] = sp2 * inv;
            bout[qg] = b;
        }
    }
}

// ---------------------------------------------------------------------------
extern "C" void kernel_launch(void* const* d_in, const int* in_sizes, int n_in,
                              void* d_out, int out_size, void* d_ws, size_t ws_size,
                              hipStream_t stream) {
    (void)in_sizes; (void)n_in; (void)out_size; (void)ws_size;
    const float* x   = (const float*)d_in[0];
    const float* pos = (const float*)d_in[1];
    // d_in[2] = batch (int32) implied by contiguous equal segments; unused.
    float* out = (float*)d_out;

    char* ws = (char*)d_ws;
    float*    p6f    = (float*)(ws);                                        // B*N*8 f32
    _Float16* p6h    = (_Float16*)(ws + 1048576);                           // B*N*8 f16
    float*    pn     = (float*)(ws + 1048576 + 524288);                     // B*N f32
    int*      fpsIdx = (int*)(ws + 1048576 + 524288 + 131072);              // B*M i32
    _Float16* qhh    = (_Float16*)(ws + 1048576 + 524288 + 131072 + 32768); // B*M*8 f16
    float*    qnn    = (float*)(ws + 1048576 + 524288 + 131072 + 32768 + 131072); // B*M f32

    pack_kernel<<<(BATCH * NPTS + 255) / 256, 256, 0, stream>>>(x, pos, p6f, p6h, pn);
    fps_kernel<<<BATCH, 1024, 0, stream>>>(p6f, fpsIdx);
    gatherq_kernel<<<(BATCH * MSAMP + 255) / 256, 256, 0, stream>>>(p6f, fpsIdx, qhh, qnn);

    dim3 grid(MSAMP / QT, BATCH);
    nbr_kernel<<<grid, 256, 0, stream>>>(p6h, pn, qhh, qnn, x, pos, out);
}
